// SelfAttention_13606456393866
// MI455X (gfx1250) — matrix operand order repaired and checked by reading, not compiled, not run on hardware
//
#include <hip/hip_runtime.h>
#include <hip/hip_bf16.h>

// B=4, S=4096, D=256 single-head attention, fp32 in/out.
// bf16 WMMA (v_wmma_f32_16x16x32_bf16) for all GEMM stages, flash-attention
// online softmax, V stored transposed and W pre-transposed so every WMMA
// fragment load is a contiguous b128. ~75 GFLOP vs ~34MB HBM => compute-bound.

#define DD   256
#define SS   4096
#define BB   4
#define LOG2E 1.44269504088896340736f

typedef __attribute__((ext_vector_type(16))) __bf16 v16bf;
typedef __attribute__((ext_vector_type(8)))  float  v8f;
typedef __attribute__((ext_vector_type(4)))  unsigned int u32x4;

__device__ __forceinline__ unsigned short f2bf_bits(float x) {
    __bf16 b = (__bf16)x;
    return __builtin_bit_cast(unsigned short, b);
}

// ---------------------------------------------------------------------------
// Kernel 0: transpose + bf16-convert the three 256x256 weight matrices.
// Wt[sel][d_out][d_in] = W[sel][d_in][d_out].  192K elements, one-shot.
// ---------------------------------------------------------------------------
__global__ __launch_bounds__(256, 1)
void wt_transpose_kernel(const float* __restrict__ Wq,
                         const float* __restrict__ Wk,
                         const float* __restrict__ Wv,
                         __bf16* __restrict__ Wtb) {
    int idx = blockIdx.x * 256 + threadIdx.x;        // 0 .. 3*65536-1
    int sel = idx >> 16;
    int r   = idx & 65535;
    int o   = r >> 8;                                // d_out
    int i   = r & 255;                               // d_in
    const float* W = (sel == 0) ? Wq : (sel == 1) ? Wk : Wv;
    Wtb[idx] = (__bf16)W[i * DD + o];                // contiguous writes
}

// ---------------------------------------------------------------------------
// Kernel 1: QKV projection.  X[16384,256] f32 * W -> bf16.
// One wave per 16x16 tile, 8 x wmma_f32_16x16x32_bf16 over K=256.
// sel 0 -> Q (scaled 1/16), 1 -> K, 2 -> V (stored transposed [B,D,S]).
// B-fragments come from the pre-transposed bf16 weights: contiguous 32B/lane.
// ---------------------------------------------------------------------------
__global__ __launch_bounds__(256, 1)
void qkv_proj_kernel(const float* __restrict__ X,
                     const __bf16* __restrict__ Wtb,
                     __bf16* __restrict__ Qb,
                     __bf16* __restrict__ Kb,
                     __bf16* __restrict__ Vt) {
    const int lane = threadIdx.x & 31;
    const int wave = threadIdx.x >> 5;
    const int tile = blockIdx.x * 8 + wave;          // 49152 tiles total
    const int sel  = tile / (1024 * 16);             // 0=Q,1=K,2=V
    const int rem  = tile % (1024 * 16);
    const int rt   = rem / 16;                       // row tile (16 rows of X)
    const int ct   = rem % 16;                       // col tile (16 cols of W)
    const int m    = lane & 15;
    const int h    = lane >> 4;

    const __bf16* Wt = Wtb + (size_t)sel * DD * DD;  // [d_out][d_in]

    v8f c = {};
    #pragma unroll
    for (int kk = 0; kk < 8; ++kk) {
        // A fragment (16x32 bf16): lane = row m; k = e + 16h (contiguous)
        const float* xp = X + (size_t)(rt * 16 + m) * DD + kk * 32;
        v16bf a;
        #pragma unroll
        for (int e = 0; e < 8; ++e) {
            a[e]     = (__bf16)xp[e + 16 * h];
            a[e + 8] = (__bf16)xp[e + 8 + 16 * h];
        }
        // B fragment (32x16 bf16): lane = col n=m; k = e + 16h, contiguous in Wt
        v16bf bm = *(const v16bf*)(Wt + (size_t)(ct * 16 + m) * DD + kk * 32 + 16 * h);

        c = __builtin_amdgcn_wmma_f32_16x16x32_bf16(false, a, false, bm,
                                                    (short)0, c, false, false);
    }

    // C/D layout: value v -> row rt*16 + v + 8h, col ct*16 + m
    if (sel == 2) {
        #pragma unroll
        for (int v = 0; v < 8; ++v) {
            int r = rt * 16 + v + 8 * h;             // global row = b*S + s
            int b = r >> 12, s = r & 4095;
            int d = ct * 16 + m;
            Vt[((size_t)b * DD + d) * SS + s] = (__bf16)c[v];
        }
    } else {
        const float sc = (sel == 0) ? 0.0625f : 1.0f; // fold 1/sqrt(256) into Q
        __bf16* O = (sel == 0) ? Qb : Kb;
        #pragma unroll
        for (int v = 0; v < 8; ++v) {
            int r = rt * 16 + v + 8 * h;
            O[(size_t)r * DD + ct * 16 + m] = (__bf16)(c[v] * sc);
        }
    }
}

// ---------------------------------------------------------------------------
// Kernel 2: flash attention.  One wave owns 16 query rows x full D=256.
// Per 32-wide KV tile: S = Q*K^T (16 wmma), online softmax (shfl_xor + v_exp),
// per-wave LDS round-trip (C-layout -> A-layout bf16 P), O += P*V (16 wmma),
// plus global_prefetch of the next KV tile.
// ---------------------------------------------------------------------------
__global__ __launch_bounds__(256, 1)
void flash_attn_kernel(const __bf16* __restrict__ Qb,
                       const __bf16* __restrict__ Kb,
                       const __bf16* __restrict__ Vt,
                       float* __restrict__ Out) {
    __shared__ __align__(16) unsigned short Plds[8][16][32]; // 1KB per wave

    const int lane = threadIdx.x & 31;
    const int wave = threadIdx.x >> 5;
    const int w    = blockIdx.x * 8 + wave;          // 1024 waves total
    const int b    = w >> 8;                         // batch
    const int qb_  = (w & 255) * 16;                 // query row base
    const int m    = lane & 15;
    const int h    = lane >> 4;

    // Preload Q as 8 A-fragments (16x32): two contiguous 16B chunks per lane.
    v16bf qf[8];
    const __bf16* qrow = Qb + ((size_t)b * SS + qb_ + m) * DD;
    #pragma unroll
    for (int kk = 0; kk < 8; ++kk) {
        union { u32x4 u[2]; v16bf v; } uu;
        uu.u[0] = *(const u32x4*)(qrow + kk * 32 + 16 * h);
        uu.u[1] = *(const u32x4*)(qrow + kk * 32 + 8 + 16 * h);
        qf[kk] = uu.v;
    }

    v8f o[16];
    #pragma unroll
    for (int t = 0; t < 16; ++t) o[t] = (v8f){};
    float mrun[8], lrun[8];
    #pragma unroll
    for (int v = 0; v < 8; ++v) { mrun[v] = -1e30f; lrun[v] = 0.0f; }

    const __bf16* kp_base = Kb + (size_t)b * SS * DD;
    const __bf16* vp_base = Vt + (size_t)b * DD * SS;

    for (int j = 0; j < SS; j += 32) {
        // prefetch next KV tile (global_prefetch_b8, CDNA5)
        if (j + 32 < SS) {
            __builtin_prefetch((const void*)(kp_base + (size_t)(j + 32 + lane) * DD), 0, 0);
            __builtin_prefetch((const void*)(vp_base + ((size_t)lane << 3) * SS + j + 32), 0, 0);
        }

        // ---- S tile: Q(16x256) x K_j^T -> two 16x16 f32 tiles ----
        v8f c0 = {}, c1 = {};
        #pragma unroll
        for (int kk = 0; kk < 8; ++kk) {
            // B fragment (32x16): lane = kv col n, k = d = kk*32 + 16h + e (contig)
            v16bf b0 = *(const v16bf*)(kp_base + (size_t)(j + m) * DD + kk * 32 + 16 * h);
            v16bf b1 = *(const v16bf*)(kp_base + (size_t)(j + 16 + m) * DD + kk * 32 + 16 * h);
            c0 = __builtin_amdgcn_wmma_f32_16x16x32_bf16(false, qf[kk], false, b0,
                                                         (short)0, c0, false, false);
            c1 = __builtin_amdgcn_wmma_f32_16x16x32_bf16(false, qf[kk], false, b1,
                                                         (short)0, c1, false, false);
        }

        // ---- online softmax; row of value v is (v + 8h), cols = m / m+16 ----
        #pragma unroll
        for (int v = 0; v < 8; ++v) {
            float t = fmaxf(c0[v], c1[v]);
            t = fmaxf(t, __shfl_xor(t, 1, 32));
            t = fmaxf(t, __shfl_xor(t, 2, 32));
            t = fmaxf(t, __shfl_xor(t, 4, 32));
            t = fmaxf(t, __shfl_xor(t, 8, 32));
            float mnew = fmaxf(mrun[v], t);
            float corr = __builtin_amdgcn_exp2f((mrun[v] - mnew) * LOG2E);
            float p0 = __builtin_amdgcn_exp2f((c0[v] - mnew) * LOG2E);
            float p1 = __builtin_amdgcn_exp2f((c1[v] - mnew) * LOG2E);
            float rs = p0 + p1;
            rs += __shfl_xor(rs, 1, 32);
            rs += __shfl_xor(rs, 2, 32);
            rs += __shfl_xor(rs, 4, 32);
            rs += __shfl_xor(rs, 8, 32);
            lrun[v] = lrun[v] * corr + rs;
            mrun[v] = mnew;

            // park P in LDS (C layout), bf16
            int row = v + 8 * h;
            Plds[wave][row][m]      = f2bf_bits(p0);
            Plds[wave][row][m + 16] = f2bf_bits(p1);

            // rescale accumulators for this row
            #pragma unroll
            for (int t2 = 0; t2 < 16; ++t2) o[t2][v] *= corr;
        }

        // ---- re-read P as A-fragment (same-wave DS ops are in-order) ----
        union { u32x4 u[2]; v16bf v; } pu;
        pu.u[0] = *(const u32x4*)&Plds[wave][m][16 * h];
        pu.u[1] = *(const u32x4*)&Plds[wave][m][8 + 16 * h];
        v16bf pa = pu.v;

        // ---- O += P(16x32) x V(32x256): 16 wmma, contiguous V loads ----
        #pragma unroll
        for (int t = 0; t < 16; ++t) {
            // B fragment: lane = d col n, k = kv = j + 16h + e (contiguous in Vt)
            v16bf bv = *(const v16bf*)(vp_base + (size_t)(t * 16 + m) * SS + j + 16 * h);
            o[t] = __builtin_amdgcn_wmma_f32_16x16x32_bf16(false, pa, false, bv,
                                                           (short)0, o[t], false, false);
        }
    }

    // ---- epilogue: O / l -> fp32 out ----
    float linv[8];
    #pragma unroll
    for (int v = 0; v < 8; ++v) linv[v] = 1.0f / lrun[v];
    float* orow = Out + ((size_t)b * SS + qb_) * DD;
    #pragma unroll
    for (int t = 0; t < 16; ++t) {
        #pragma unroll
        for (int v = 0; v < 8; ++v)
            orow[(size_t)(v + 8 * h) * DD + t * 16 + m] = o[t][v] * linv[v];
    }
}

// ---------------------------------------------------------------------------
extern "C" void kernel_launch(void* const* d_in, const int* in_sizes, int n_in,
                              void* d_out, int out_size, void* d_ws, size_t ws_size,
                              hipStream_t stream) {
    const float* X  = (const float*)d_in[0];
    const float* Wq = (const float*)d_in[1];
    const float* Wk = (const float*)d_in[2];
    const float* Wv = (const float*)d_in[3];
    float* out = (float*)d_out;

    // workspace layout (bf16):
    //   Qb, Kb row-major [B,S,D]; Vt transposed [B,D,S]; Wtb [3][D_out][D_in]
    const size_t elems = (size_t)BB * SS * DD;       // 4M elements each
    __bf16* Qb  = (__bf16*)d_ws;
    __bf16* Kb  = Qb + elems;
    __bf16* Vt  = Kb + elems;
    __bf16* Wtb = Vt + elems;                        // 3*64K elems = 384KB

    // 3*256*256 elements / 256 threads -> 768 blocks
    wt_transpose_kernel<<<768, 256, 0, stream>>>(Wq, Wk, Wv, Wtb);
    // 3 * 1024 * 16 = 49152 tiles, 8 waves/block -> 6144 blocks
    qkv_proj_kernel<<<6144, 256, 0, stream>>>(X, Wtb, Qb, Kb, Vt);
    // 4 * 256 = 1024 waves, 8 waves/block -> 128 blocks
    flash_attn_kernel<<<128, 256, 0, stream>>>(Qb, Kb, Vt, out);
}
